// CausalSelfAttention_42923903156761
// MI455X (gfx1250) — matrix operand order repaired and checked by reading, not compiled, not run on hardware
//
#include <hip/hip_runtime.h>
#include <hip/hip_bf16.h>
#include <math.h>

// ---------------------------------------------------------------------------
// CDNA5 (gfx1250) causal self-attention: bf16 WMMA, f32 accumulate,
// flash-attention tiling, TDM (tensor_load_to_lds) staging of K/V panels.
// B=2, T=2048, C=1024, H=16, D=64.
// ---------------------------------------------------------------------------

typedef __attribute__((ext_vector_type(16))) __bf16 v16bf;
typedef __attribute__((ext_vector_type(8)))  __bf16 v8bf;
typedef __attribute__((ext_vector_type(8)))  float  v8f;

typedef unsigned tdm_u4 __attribute__((ext_vector_type(4)));
typedef int      tdm_i4 __attribute__((ext_vector_type(4)));
typedef int      tdm_i8 __attribute__((ext_vector_type(8)));

static constexpr int Bn = 2;
static constexpr int Tn = 2048;
static constexpr int Cn = 1024;
static constexpr int Hn = 16;
static constexpr int Dn = 64;

// ---- helpers --------------------------------------------------------------

__device__ __forceinline__ unsigned short f2bf(float f) {
  unsigned u = __builtin_bit_cast(unsigned, f);
  u += 0x7FFFu + ((u >> 16) & 1u);            // round-to-nearest-even
  return (unsigned short)(u >> 16);
}

__device__ __forceinline__ v8bf ld8(const unsigned short* p) {
  return *(const v8bf*)(const void*)p;        // 16B -> global_load_b128 / ds_load_b128
}

__device__ __forceinline__ v16bf cat16(v8bf lo, v8bf hi) {
  return __builtin_shufflevector(lo, hi, 0,1,2,3,4,5,6,7,8,9,10,11,12,13,14,15);
}

__device__ __forceinline__ v8f wmma_bf16(v16bf a, v16bf b, v8f c) {
  // (neg_a, A, neg_b, B, c_mod, C, reuse_a, reuse_b)
  return __builtin_amdgcn_wmma_f32_16x16x32_bf16(false, a, false, b, (short)0, c,
                                                 false, false);
}

__device__ __forceinline__ float rowmax16(float v) {
  #pragma unroll
  for (int m = 1; m < 16; m <<= 1) v = fmaxf(v, __shfl_xor(v, m, 32));
  return v;   // xor of bits 0..3 stays inside each 16-lane half
}

__device__ __forceinline__ float rowsum16(float v) {
  #pragma unroll
  for (int m = 1; m < 16; m <<= 1) v += __shfl_xor(v, m, 32);
  return v;
}

// A-fragment (16x32 bf16, ISA 7.12.2): lane = half*16+lo holds row M=lo,
//   elems 0..7  -> K = kbase + half*8 + i        (contiguous 8)
//   elems 8..15 -> K = kbase + 16 + half*8 + i-8 (contiguous 8)
__device__ __forceinline__ v16bf load_afrag(const unsigned short* rowp, int kbase, int hl) {
  return cat16(ld8(rowp + kbase + hl * 8), ld8(rowp + kbase + 16 + hl * 8));
}
// B-fragment (32x16 bf16): lane = half*16+lo holds col N=lo,
//   elems 0..7  -> K = kbase + half*8 + i        (contiguous 8)
//   elems 8..15 -> K = kbase + 16 + half*8 + i-8 (contiguous 8)
__device__ __forceinline__ v16bf load_bfrag(const unsigned short* colp, int kbase, int hl) {
  return cat16(ld8(colp + kbase + hl * 8), ld8(colp + kbase + 16 + hl * 8));
}

// ---- Tensor Data Mover: 2D tile (16-bit elements) Global -> LDS -----------
// Builds a D# (group0 128b + group1 256b) per CDNA5 ISA and
// issues TENSOR_LOAD_TO_LDS. Tracked by TENSORcnt.
#if __has_builtin(__builtin_amdgcn_tensor_load_to_lds)
#define HAVE_TDM 1
#else
#define HAVE_TDM 0
#endif

#if HAVE_TDM
__device__ __forceinline__ void tdm_load_2d_b16(unsigned lds_off, const void* gaddr,
                                                unsigned tile_w, unsigned tile_h,
                                                unsigned tensor_w, unsigned tensor_h,
                                                unsigned row_stride /*elements*/) {
  unsigned long long ga = (unsigned long long)gaddr;
  tdm_u4 g0;
  g0[0] = 1u;                                   // count=1 (valid), user mode
  g0[1] = lds_off;                              // LDS byte address
  g0[2] = (unsigned)ga;                         // global_addr[31:0]
  g0[3] = ((unsigned)(ga >> 32) & 0x01FFFFFFu)  // global_addr[56:32]
          | (2u << 30);                         // type = 2 ("image")
  tdm_i8 g1;
  g1[0] = (int)(1u << 16);                      // workgroup_mask=0, data_size=1 (2B)
  g1[1] = (int)(tensor_w << 16);                // tensor_dim0[15:0] @ bits 63:48
  g1[2] = (int)((tensor_w >> 16) | (tensor_h << 16));   // dim0 hi | dim1 lo
  g1[3] = (int)((tensor_h >> 16) | (tile_w << 16));     // dim1 hi | tile_dim0
  g1[4] = (int)tile_h;                          // tile_dim1 (tile_dim2 = 0)
  g1[5] = (int)row_stride;                      // tensor_dim0_stride[31:0]
  g1[6] = 0;                                    // stride0 hi | stride1 lo
  g1[7] = 0;                                    // stride1 hi
  tdm_i4 gz = {0, 0, 0, 0};
#if __clang_major__ >= 23
  tdm_i8 gz8 = {0, 0, 0, 0, 0, 0, 0, 0};
  __builtin_amdgcn_tensor_load_to_lds(g0, g1, gz, gz, gz8, 0);
#else
  __builtin_amdgcn_tensor_load_to_lds(g0, g1, gz, gz, 0);
#endif
}
#endif

// ---- precision conversion kernels -----------------------------------------

__global__ void cvt_f32_to_bf16(const float* __restrict__ src,
                                unsigned short* __restrict__ dst, int n) {
  int i = blockIdx.x * blockDim.x + threadIdx.x;
  if (i < n) dst[i] = f2bf(src[i]);
}

// dst[n*K + k] = src[k*N + n]  (transpose so GEMM B-fragments load contiguous K)
__global__ void cvtT_f32_to_bf16(const float* __restrict__ src,
                                 unsigned short* __restrict__ dst, int K, int N) {
  int i = blockIdx.x * blockDim.x + threadIdx.x;
  if (i < K * N) {
    int n = i / K, k = i % K;
    dst[i] = f2bf(src[k * N + n]);
  }
}

// ---- QKV projection: 32x64 register tile per wave (8 WMMA / k-step) -------
__global__ void __launch_bounds__(128)
qkv_gemm(const unsigned short* __restrict__ xb,
         const unsigned short* __restrict__ wT,      // [3C][C] transposed
         const float* __restrict__ bias,
         unsigned short* __restrict__ Qb,            // [B,H,T,D]
         unsigned short* __restrict__ Kb,            // [B,H,T,D]
         unsigned short* __restrict__ Vt) {          // [B,H,D,T]
  const int NT = (3 * Cn) / 64;                      // 48 tiles along N
  int wave = threadIdx.x >> 5, lane = threadIdx.x & 31;
  int tile = blockIdx.x * 4 + wave;
  int row0 = (tile / NT) * 32, col0 = (tile % NT) * 64;
  int hl = lane >> 4, lo = lane & 15;

  const unsigned short* ar[2];
  const unsigned short* br[4];
  #pragma unroll
  for (int m = 0; m < 2; m++) ar[m] = xb + (size_t)(row0 + m * 16 + lo) * Cn;
  #pragma unroll
  for (int n = 0; n < 4; n++) br[n] = wT + (size_t)(col0 + n * 16 + lo) * Cn;

  v8f acc[2][4];
  #pragma unroll
  for (int m = 0; m < 2; m++)
    #pragma unroll
    for (int n = 0; n < 4; n++) acc[m][n] = (v8f){0, 0, 0, 0, 0, 0, 0, 0};

  for (int kb = 0; kb < Cn; kb += 32) {
    __builtin_prefetch(ar[0] + kb + 128, 0, 0);      // global_prefetch_b8
    v16bf a[2], b[4];
    #pragma unroll
    for (int m = 0; m < 2; m++) a[m] = load_afrag(ar[m], kb, hl);
    #pragma unroll
    for (int n = 0; n < 4; n++) b[n] = load_bfrag(br[n], kb, hl);
    #pragma unroll
    for (int m = 0; m < 2; m++)
      #pragma unroll
      for (int n = 0; n < 4; n++) acc[m][n] = wmma_bf16(a[m], b[n], acc[m][n]);
  }

  #pragma unroll
  for (int n = 0; n < 4; n++) {
    int colg = col0 + n * 16 + lo;
    float bv = bias[colg];
    int seg = colg >> 10;                            // 0:q 1:k 2:v (64-wide tile never straddles)
    int cc = colg & (Cn - 1);
    int h = cc >> 6, d = cc & (Dn - 1);
    #pragma unroll
    for (int m = 0; m < 2; m++) {
      #pragma unroll
      for (int j = 0; j < 8; j++) {                  // C-layout row = +j + hl*8
        int row = row0 + m * 16 + j + hl * 8;
        int bb = row >> 11, t = row & (Tn - 1);
        unsigned short val = f2bf(acc[m][n][j] + bv);
        if (seg == 0)      Qb[((size_t)(bb * Hn + h) * Tn + t) * Dn + d] = val;
        else if (seg == 1) Kb[((size_t)(bb * Hn + h) * Tn + t) * Dn + d] = val;
        else               Vt[((size_t)(bb * Hn + h) * Dn + d) * Tn + t] = val;
      }
    }
  }
}

// ---- flash attention: one wave per (b,h,16 query rows) --------------------
__global__ void __launch_bounds__(32)
attn_kernel(const unsigned short* __restrict__ Qb,
            const unsigned short* __restrict__ Kb,
            const unsigned short* __restrict__ Vt,   // [B,H,D,T]
            unsigned short* __restrict__ Yb) {       // [B,T,C] bf16
  __shared__ unsigned short kpan[32 * 64];           // K panel [key][d]   (4 KB)
  __shared__ unsigned short vpan[64 * 32];           // V panel [d][key]   (4 KB)
  __shared__ unsigned short pbuf[16 * 32];           // P re-layout C->A   (1 KB)

  int id = blockIdx.x;
  int qt = id & (Tn / 16 - 1);
  int bh = id >> 7;
  int h = bh & (Hn - 1), b = bh >> 4;
  int q0 = qt * 16;
  int lane = threadIdx.x & 31, hl = lane >> 4, lo = lane & 15;

  const size_t head = (size_t)(b * Hn + h) * Tn * Dn;
  const unsigned short* qrow = Qb + head + (size_t)(q0 + lo) * Dn;
  v16bf aq0 = load_afrag(qrow, 0, hl);               // d = 0..31
  v16bf aq1 = load_afrag(qrow, 32, hl);              // d = 32..63

  v8f o[4] = {{0,0,0,0,0,0,0,0}, {0,0,0,0,0,0,0,0},
              {0,0,0,0,0,0,0,0}, {0,0,0,0,0,0,0,0}};
  float rm[8], rl[8];
  #pragma unroll
  for (int j = 0; j < 8; j++) { rm[j] = -__builtin_inff(); rl[j] = 0.f; }

  const unsigned short* kbase = Kb + head;
  const unsigned short* vbase = Vt + head;           // [D][T]

#if HAVE_TDM
  unsigned kpan_off = (unsigned)(unsigned long long)(void*)kpan;
  unsigned vpan_off = (unsigned)(unsigned long long)(void*)vpan;
#endif

  const int nsteps = (q0 + 15) / 32 + 1;             // causal horizon, 32 keys/step
  for (int s = 0; s < nsteps; s++) {
    int k0 = s * 32;

    // ---- stage K (32x64) and V (64x32) panels into LDS via TDM ----
#if HAVE_TDM
    tdm_load_2d_b16(kpan_off, kbase + (size_t)k0 * Dn,
                    /*tile*/ Dn, 32, /*tensor*/ Dn, (unsigned)(Tn - k0),
                    /*stride*/ Dn);
    tdm_load_2d_b16(vpan_off, vbase + k0,
                    /*tile*/ 32, Dn, /*tensor*/ (unsigned)(Tn - k0), Dn,
                    /*stride*/ Tn);
    __builtin_amdgcn_s_wait_tensorcnt(0);
#else
    // fallback: cooperative copy (K panel is contiguous 4 KB)
    for (int i = lane; i < (32 * Dn) / 8; i += 32)
      ((v8bf*)kpan)[i] = ld8(kbase + (size_t)k0 * Dn + i * 8);
    for (int r = lane; r < Dn; r += 32)
      #pragma unroll
      for (int c2 = 0; c2 < 4; c2++)
        ((v8bf*)(vpan + r * 32))[c2] = ld8(vbase + (size_t)r * Tn + k0 + c2 * 8);
#endif
    __syncthreads();

    // ---- S = Q K^T over 32 keys: two 16x16 C-tiles, 2 WMMA each ----
    v8f st[2];
    #pragma unroll
    for (int t = 0; t < 2; t++) {
      const unsigned short* krow = kpan + (t * 16 + lo) * Dn;
      v16bf bk0 = load_bfrag(krow, 0, hl);           // ds_load_b128
      v16bf bk1 = load_bfrag(krow, 32, hl);
      v8f z = {0, 0, 0, 0, 0, 0, 0, 0};
      v8f sc = wmma_bf16(aq0, bk0, z);
      st[t] = wmma_bf16(aq1, bk1, sc);
    }

    // ---- online softmax (f32) ----
    #pragma unroll
    for (int j = 0; j < 8; j++) {
      int row = q0 + j + hl * 8;
      #pragma unroll
      for (int t = 0; t < 2; t++) {
        int col = k0 + t * 16 + lo;
        float v = st[t][j] * 0.125f;                 // 1/sqrt(64)
        st[t][j] = (col <= row) ? v : -__builtin_inff();
      }
      float mx = rowmax16(fmaxf(st[0][j], st[1][j]));
      float mn = fmaxf(rm[j], mx);
      float alpha = __expf(rm[j] - mn);              // 0 on first step (rm=-inf)
      float p0 = __expf(st[0][j] - mn);
      float p1 = __expf(st[1][j] - mn);
      rl[j] = rl[j] * alpha + rowsum16(p0 + p1);
      rm[j] = mn;
      #pragma unroll
      for (int c2 = 0; c2 < 4; c2++) o[c2][j] *= alpha;
      pbuf[(j + hl * 8) * 32 + lo]      = f2bf(p0);  // stage P row-major in LDS
      pbuf[(j + hl * 8) * 32 + 16 + lo] = f2bf(p1);
    }
    __syncthreads();                                 // single wave: uniform
    v16bf ap = load_afrag(pbuf + lo * 32, 0, hl);    // ds_load_b128 x2
    __syncthreads();

    // ---- O += P @ V : 4 chunks of 16 d-columns ----
    #pragma unroll
    for (int c2 = 0; c2 < 4; c2++) {
      const unsigned short* vrow = vpan + (c2 * 16 + lo) * 32;
      v16bf bv = load_bfrag(vrow, 0, hl);
      o[c2] = wmma_bf16(ap, bv, o[c2]);
    }
    __syncthreads();                                 // panels reused next step
  }

  // normalize + write y[B,T,C] as bf16 for the projection GEMM
  #pragma unroll
  for (int j = 0; j < 8; j++) {
    float inv = 1.0f / rl[j];
    int q = q0 + j + hl * 8;
    size_t base = ((size_t)(b * Tn + q)) * Cn + h * Dn;
    #pragma unroll
    for (int c2 = 0; c2 < 4; c2++)
      Yb[base + c2 * 16 + lo] = f2bf(o[c2][j] * inv);
  }
}

// ---- output projection: 32x64 register tile, f32 out ----------------------
__global__ void __launch_bounds__(128)
proj_gemm(const unsigned short* __restrict__ yb,
          const unsigned short* __restrict__ wT,     // [C][C] transposed
          const float* __restrict__ bias,
          float* __restrict__ out) {
  const int NT = Cn / 64;                            // 16
  int wave = threadIdx.x >> 5, lane = threadIdx.x & 31;
  int tile = blockIdx.x * 4 + wave;
  int row0 = (tile / NT) * 32, col0 = (tile % NT) * 64;
  int hl = lane >> 4, lo = lane & 15;

  const unsigned short* ar[2];
  const unsigned short* br[4];
  #pragma unroll
  for (int m = 0; m < 2; m++) ar[m] = yb + (size_t)(row0 + m * 16 + lo) * Cn;
  #pragma unroll
  for (int n = 0; n < 4; n++) br[n] = wT + (size_t)(col0 + n * 16 + lo) * Cn;

  v8f acc[2][4];
  #pragma unroll
  for (int m = 0; m < 2; m++)
    #pragma unroll
    for (int n = 0; n < 4; n++) acc[m][n] = (v8f){0, 0, 0, 0, 0, 0, 0, 0};

  for (int kb = 0; kb < Cn; kb += 32) {
    __builtin_prefetch(ar[0] + kb + 128, 0, 0);
    v16bf a[2], b[4];
    #pragma unroll
    for (int m = 0; m < 2; m++) a[m] = load_afrag(ar[m], kb, hl);
    #pragma unroll
    for (int n = 0; n < 4; n++) b[n] = load_bfrag(br[n], kb, hl);
    #pragma unroll
    for (int m = 0; m < 2; m++)
      #pragma unroll
      for (int n = 0; n < 4; n++) acc[m][n] = wmma_bf16(a[m], b[n], acc[m][n]);
  }

  #pragma unroll
  for (int n = 0; n < 4; n++) {
    float bv = bias[col0 + n * 16 + lo];
    #pragma unroll
    for (int m = 0; m < 2; m++) {
      #pragma unroll
      for (int j = 0; j < 8; j++) {
        int row = row0 + m * 16 + j + hl * 8;
        out[(size_t)row * Cn + col0 + n * 16 + lo] = acc[m][n][j] + bv;
      }
    }
  }
}

// ---- host launcher --------------------------------------------------------

extern "C" void kernel_launch(void* const* d_in, const int* in_sizes, int n_in,
                              void* d_out, int out_size, void* d_ws, size_t ws_size,
                              hipStream_t stream) {
  (void)in_sizes; (void)n_in; (void)out_size; (void)ws_size;
  const float* x     = (const float*)d_in[0];
  const float* Wqkv  = (const float*)d_in[1];
  const float* bqkv  = (const float*)d_in[2];
  const float* Wproj = (const float*)d_in[3];
  const float* bproj = (const float*)d_in[4];
  float* out = (float*)d_out;

  char* ws = (char*)d_ws;
  size_t off = 0;
  auto alloc = [&](size_t bytes) -> void* {
    void* p = ws + off;
    off += (bytes + 255) & ~(size_t)255;
    return p;
  };

  const size_t BT = (size_t)Bn * Tn;                     // 4096 tokens
  unsigned short* xb     = (unsigned short*)alloc(BT * Cn * 2);             // 8 MB
  unsigned short* wqkvT  = (unsigned short*)alloc((size_t)3 * Cn * Cn * 2); // 6 MB
  unsigned short* wprojT = (unsigned short*)alloc((size_t)Cn * Cn * 2);     // 2 MB
  unsigned short* Qb     = (unsigned short*)alloc(BT * Cn * 2);             // 8 MB
  unsigned short* Kb     = (unsigned short*)alloc(BT * Cn * 2);             // 8 MB
  unsigned short* Vt     = (unsigned short*)alloc(BT * Cn * 2);             // 8 MB
  unsigned short* Yb     = (unsigned short*)alloc(BT * Cn * 2);             // 8 MB

  int nx = (int)(BT * Cn);
  cvt_f32_to_bf16<<<(nx + 255) / 256, 256, 0, stream>>>(x, xb, nx);
  cvtT_f32_to_bf16<<<((3 * Cn * Cn) + 255) / 256, 256, 0, stream>>>(Wqkv, wqkvT, Cn, 3 * Cn);
  cvtT_f32_to_bf16<<<((Cn * Cn) + 255) / 256, 256, 0, stream>>>(Wproj, wprojT, Cn, Cn);

  // 32x64 wave tiles: (4096/32)*(3072/64) = 128*48 = 6144 waves, 4/block
  qkv_gemm<<<6144 / 4, 128, 0, stream>>>(xb, wqkvT, bqkv, Qb, Kb, Vt);

  attn_kernel<<<Bn * Hn * (Tn / 16), 32, 0, stream>>>(Qb, Kb, Vt, Yb);

  // (4096/32)*(1024/64) = 128*16 = 2048 waves, 4/block
  proj_gemm<<<2048 / 4, 128, 0, stream>>>(Yb, wprojT, bproj, out);
}
